// HybridMambaLogisticModelPro_51067161150072
// MI455X (gfx1250) — compile-verified, matrix operand order checked by reading, not run on hardware
//
#include <hip/hip_runtime.h>
#include <hip/hip_bf16.h>
#include <math.h>

// ---------------------------------------------------------------------------
// HybridMambaLogisticModelPro — MI455X (gfx1250) implementation.
// L == 1 collapses the Mamba scan:  y = softplus(dt@Wdt+b) * u * <B,C> + u*D,
// gated by silu(z).  A_log is dead (h0 == 0).  The three big GEMMs
// (512->3072, 1536->96, 1536->512 over batch 1024) run on bf16 WMMA with the
// shared B tile staged through LDS via global_load_async_to_lds_b128
// (ASYNCcnt double-buffering) and sched_barrier-pinned pipelined fragments.
// ---------------------------------------------------------------------------

typedef __bf16 bf16_t;
typedef __attribute__((ext_vector_type(16))) __bf16 v16bf;
typedef __attribute__((ext_vector_type(8)))  __bf16 v8bf;
typedef __attribute__((ext_vector_type(8)))  float  v8f;

#define N_LAYERS 4
#define D_MODEL  512
#define D_INNER  1536
#define D_STATE  32
#define DT_RANK  32
#define XDBL     96      // DT_RANK + 2*D_STATE
#define BATCH    1024
#define N_FEAT   68
#define N_RAIN   4
#define N_NONRAIN 64

static __device__ __forceinline__ float sigf(float x) { return 1.0f / (1.0f + __expf(-x)); }
static __device__ __forceinline__ float siluf(float x){ return x * sigf(x); }

// ---------------------------------------------------------------------------
// bf16 WMMA fragment load.  ISA 7.12.2, 16-bit A 16x32 layout:
//   lane = m + 16*h ; element e of v16bf holds k = (e&7) + 8*h + 16*(e>>3)
// i.e. two contiguous 8-element (16B) runs at k0+8h and k0+16+8h.
// B (32x16, stored as weight rows W[n, k]) mirrors this with lane = n + 16*h.
// Works for both global and LDS source pointers.
// ---------------------------------------------------------------------------
static __device__ __forceinline__ v16bf load_frag(const bf16_t* __restrict__ p, int kh)
{
    const v8bf lo = *(const v8bf*)(p + 8 * kh);
    const v8bf hi = *(const v8bf*)(p + 16 + 8 * kh);
    v16bf r;
#pragma unroll
    for (int i = 0; i < 8; ++i) { r[i] = lo[i]; r[i + 8] = hi[i]; }
    return r;
}

// CDNA5 async global->LDS copy, 16 bytes per lane (tracked by ASYNCcnt).
static __device__ __forceinline__ void async_b128(unsigned int lds_byte_off, const void* gaddr)
{
    asm volatile("global_load_async_to_lds_b128 %0, %1, off"
                 :: "v"(lds_byte_off), "v"(gaddr) : "memory");
}

// ---------------------------------------------------------------------------
// C = A(bf16, 1024xK row-major) * W(bf16, NxK row-major)^T  [+ Cinit]
// block = 256 threads = 8 waves; wave tile = 16 rows x (NSUB*16) cols.
// K-stage = 128: the NRx128 B tile (shared by all 8 waves) is copied to LDS
// with global_load_async_to_lds_b128, double buffered (s_wait_asynccnt NSUB
// keeps the next stage in flight while the previous completes).  Fragment
// loads are pipelined one kk-step ahead and pinned with sched_barrier(0) so
// the backend cannot sink them to their uses: kk+1's loads issue above kk's
// WMMAs, forcing true double-buffered fragment registers and partial
// dscnt/loadcnt waits instead of full drains before every WMMA.
// LDS rows padded to 136 elems (272B = 17x16B) -> conflict-free b128 reads.
// Requires: K % 128 == 0, gridDim.y = 8 (M = 1024).
// ---------------------------------------------------------------------------
template<int NSUB, bool HAS_C>
__global__ __launch_bounds__(256)
void gemm_bf16_wmma(const bf16_t* __restrict__ A, const bf16_t* __restrict__ W,
                    const float* __restrict__ Cinit, float* __restrict__ C,
                    int N, int K)
{
    constexpr int NR    = NSUB * 16;      // B-tile rows (output cols per block)
    constexpr int LDSTR = 136;            // padded k-stride in elements
    __shared__ bf16_t bsh[2][NR * LDSTR];

    const int tid  = threadIdx.x;
    const int lane = tid & 31;
    const int wave = tid >> 5;
    const int kh   = lane >> 4;           // K-half select
    const int l16  = lane & 15;           // row (A) / col (B,C) within tile
    const int m0   = blockIdx.y * 128 + wave * 16;
    const int n0   = blockIdx.x * NR;

    v8f acc[NSUB];
#pragma unroll
    for (int s = 0; s < NSUB; ++s)
#pragma unroll
        for (int v = 0; v < 8; ++v)
            acc[s][v] = HAS_C
                ? Cinit[(size_t)(m0 + v + 8 * kh) * (size_t)N + (n0 + s * 16 + l16)]
                : 0.0f;

    // cooperative async B-stage: NR rows x 128 k  ==  NR*16 chunks of 16B,
    // NSUB chunks per thread.  thread -> (row = tid>>4 + 16j, kc = (tid&15)*8)
    const unsigned int lds_base = (unsigned int)(uintptr_t)&bsh[0][0];
    constexpr unsigned int BUFBYTES = (unsigned int)(NR * LDSTR * 2);
    const int rbase = tid >> 4;
    const int kc    = (tid & 15) * 8;

    auto issue_stage = [&](int k0, int b) {
        const unsigned int dbase = lds_base + (unsigned int)b * BUFBYTES;
#pragma unroll
        for (int j = 0; j < NSUB; ++j) {
            const int row = rbase + 16 * j;
            const bf16_t* src = W + (size_t)(n0 + row) * (size_t)K + (k0 + kc);
            async_b128(dbase + (unsigned int)(row * LDSTR + kc) * 2u, src);
        }
    };

    const bf16_t* __restrict__ arow = A + (size_t)(m0 + l16) * (size_t)K;
    const int nstages = K >> 7;
    int bufsel = 0;

    issue_stage(0, 0);
    for (int st = 0; st < nstages; ++st) {
        const int k0 = st << 7;

        if (st + 1 < nstages) {
            issue_stage((st + 1) << 7, bufsel ^ 1);
            __builtin_prefetch(arow + k0 + 128, 0, 3);   // next A stage, keep near
        }

        // A frag for kk=0 is LDS-independent: issue before the barrier so its
        // latency hides behind the stage handshake.
        v16bf a_cur = load_frag(arow + k0, kh);

        if (st + 1 < nstages) {
            // previous stage done when <= NSUB (next stage's) asyncs outstanding
            if constexpr (NSUB == 4) asm volatile("s_wait_asynccnt 0x4" ::: "memory");
            else                     asm volatile("s_wait_asynccnt 0x6" ::: "memory");
        } else {
            asm volatile("s_wait_asynccnt 0x0" ::: "memory");
        }
        __syncthreads();                   // B tile visible to all 8 waves

        const bf16_t* bb = &bsh[bufsel][0];
        v16bf b_cur[NSUB], b_nxt[NSUB];
        v16bf a_nxt;
#pragma unroll
        for (int s = 0; s < NSUB; ++s)
            b_cur[s] = load_frag(bb + (s * 16 + l16) * LDSTR, kh);

#pragma unroll
        for (int kk = 0; kk < 4; ++kk) {
            if (kk < 3) {                  // preload kk+1 while kk's WMMAs run
                a_nxt = load_frag(arow + k0 + (kk + 1) * 32, kh);
#pragma unroll
                for (int s = 0; s < NSUB; ++s)
                    b_nxt[s] = load_frag(bb + (s * 16 + l16) * LDSTR + (kk + 1) * 32, kh);
            }
            // Fence: kk+1's loads must be scheduled above kk's WMMAs, so the
            // wait before the WMMAs is partial (oldest frags only).
            __builtin_amdgcn_sched_barrier(0);
#pragma unroll
            for (int s = 0; s < NSUB; ++s)
                acc[s] = __builtin_amdgcn_wmma_f32_16x16x32_bf16(
                    false, a_cur, false, b_cur[s], (short)0, acc[s], false, false);
            __builtin_amdgcn_sched_barrier(0);
            if (kk < 3) {
                a_cur = a_nxt;             // SSA renaming under full unroll
#pragma unroll
                for (int s = 0; s < NSUB; ++s) b_cur[s] = b_nxt[s];
            }
        }
        __syncthreads();                   // all waves done reading bufsel
        bufsel ^= 1;
    }

#pragma unroll
    for (int s = 0; s < NSUB; ++s) {
        const int n = n0 + s * 16 + l16;
#pragma unroll
        for (int v = 0; v < 8; ++v)
            C[(size_t)(m0 + v + 8 * kh) * (size_t)N + n] = acc[s][v];
    }
}

// ---------------------------------------------------------------------------
// f32 -> bf16 weight conversion (streaming, 4 elems/thread; sizes are %4==0).
// ---------------------------------------------------------------------------
__global__ __launch_bounds__(256)
void cvt_f32_bf16(const float* __restrict__ in, bf16_t* __restrict__ out, int n)
{
    const int i = (blockIdx.x * 256 + threadIdx.x) * 4;
    if (i < n) {
        const float4 v = *(const float4*)(in + i);
        out[i + 0] = (bf16_t)v.x;
        out[i + 1] = (bf16_t)v.y;
        out[i + 2] = (bf16_t)v.z;
        out[i + 3] = (bf16_t)v.w;
    }
}

// ---------------------------------------------------------------------------
// h = relu(nonrain @ w_in^T + b_in) ; nonrain = x[:, 4:68]
// ---------------------------------------------------------------------------
__global__ __launch_bounds__(256)
void head_in(const float* __restrict__ x, const float* __restrict__ w_in,
             const float* __restrict__ b_in, float* __restrict__ h)
{
    __shared__ float xs[N_NONRAIN];
    const int b = blockIdx.x, tid = threadIdx.x;
    if (tid < N_NONRAIN) xs[tid] = x[(size_t)b * N_FEAT + N_RAIN + tid];
    __syncthreads();
    for (int e = tid; e < D_MODEL; e += 256) {
        float a = b_in[e];
        const float* __restrict__ wr = w_in + (size_t)e * N_NONRAIN;
#pragma unroll 8
        for (int k = 0; k < N_NONRAIN; ++k) a += xs[k] * wr[k];
        h[(size_t)b * D_MODEL + e] = fmaxf(a, 0.0f);
    }
}

// ---------------------------------------------------------------------------
// LayerNorm over D_MODEL=512 -> bf16.  One wave per row, wave32 shuffles.
// ---------------------------------------------------------------------------
__global__ __launch_bounds__(256)
void layernorm_bf16(const float* __restrict__ h, const float* __restrict__ g,
                    const float* __restrict__ bb, bf16_t* __restrict__ out)
{
    const int row  = blockIdx.x * 8 + (threadIdx.x >> 5);
    const int lane = threadIdx.x & 31;
    const float* __restrict__ xr = h + (size_t)row * D_MODEL;
    float vals[16], s = 0.0f, ss = 0.0f;
#pragma unroll
    for (int i = 0; i < 16; ++i) {
        const float v = xr[lane + i * 32];
        vals[i] = v; s += v; ss += v * v;
    }
#pragma unroll
    for (int m = 16; m > 0; m >>= 1) {
        s  += __shfl_xor(s,  m, 32);
        ss += __shfl_xor(ss, m, 32);
    }
    const float mu  = s * (1.0f / D_MODEL);
    const float var = ss * (1.0f / D_MODEL) - mu * mu;
    const float inv = rsqrtf(var + 1e-5f);
#pragma unroll
    for (int i = 0; i < 16; ++i) {
        const int c = lane + i * 32;
        out[(size_t)row * D_MODEL + c] = (bf16_t)((vals[i] - mu) * inv * g[c] + bb[c]);
    }
}

// ---------------------------------------------------------------------------
// L==1 depthwise conv + SiLU:  u = silu(xc * cw[d,3] + cb[d])
// ---------------------------------------------------------------------------
__global__ __launch_bounds__(256)
void conv_silu(const float* __restrict__ xz, const float* __restrict__ cw,
               const float* __restrict__ cb, float* __restrict__ u,
               bf16_t* __restrict__ ubf)
{
    const int idx = blockIdx.x * 256 + threadIdx.x;      // over BATCH*D_INNER
    const int b = idx / D_INNER, d = idx - b * D_INNER;
    const float v = xz[(size_t)b * (2 * D_INNER) + d] * cw[d * 4 + 3] + cb[d];
    const float s = siluf(v);
    u[idx]   = s;
    ubf[idx] = (bf16_t)s;
}

// ---------------------------------------------------------------------------
// Fused SSM (L==1, h0==0):
//   s[b]   = <B[b,:], C[b,:]>                     (32-wide shuffle reduce)
//   delta  = softplus(dt[b,:] @ Wdt[d,:] + bdt[d])
//   y      = (delta * u * s[b] + u * D[d]) * silu(z)   -> bf16 for out_proj
// ---------------------------------------------------------------------------
__global__ __launch_bounds__(256)
void ssm_fused(const float* __restrict__ xdbl, const float* __restrict__ Wdt,
               const float* __restrict__ bdt, const float* __restrict__ Dp,
               const float* __restrict__ u, const float* __restrict__ xz,
               bf16_t* __restrict__ ybf)
{
    __shared__ float dt_sh[DT_RANK];
    __shared__ float s_sh;
    const int b = blockIdx.y, tid = threadIdx.x;
    const float* __restrict__ row = xdbl + (size_t)b * XDBL;
    if (tid < 32) {
        dt_sh[tid] = row[tid];
        float p = row[DT_RANK + tid] * row[DT_RANK + D_STATE + tid];
#pragma unroll
        for (int m = 16; m > 0; m >>= 1) p += __shfl_xor(p, m, 32);
        if (tid == 0) s_sh = p;
    }
    __syncthreads();

    const int d = blockIdx.x * 256 + tid;
    float a = bdt[d];
    const float* __restrict__ wr = Wdt + (size_t)d * DT_RANK;
#pragma unroll 8
    for (int k = 0; k < DT_RANK; ++k) a += dt_sh[k] * wr[k];
    const float delta = (a > 20.0f) ? a : log1pf(__expf(a));
    const float uu = u[(size_t)b * D_INNER + d];
    float y = delta * uu * s_sh + uu * Dp[d];
    const float zv = xz[(size_t)b * (2 * D_INNER) + D_INNER + d];
    y *= siluf(zv);
    ybf[(size_t)b * D_INNER + d] = (bf16_t)y;
}

// ---------------------------------------------------------------------------
// Output head: lg = sigmoid(rain @ w_log^T + b_log);
// fused = h * (lg * W_gate + b_gate); z1 = relu(fused @ W_h1^T + b_h1);
// out = sigmoid(z1 @ W_h2^T + b_h2).  One block per sample.
// ---------------------------------------------------------------------------
__global__ __launch_bounds__(256)
void head_out(const float* __restrict__ h, const float* __restrict__ x,
              const float* __restrict__ w_log, const float* __restrict__ b_log,
              const float* __restrict__ W_gate, const float* __restrict__ b_gate,
              const float* __restrict__ W_h1, const float* __restrict__ b_h1,
              const float* __restrict__ W_h2, const float* __restrict__ b_h2,
              float* __restrict__ out)
{
    __shared__ float fused[D_MODEL];
    __shared__ float z1[32];
    __shared__ float lg_sh;
    const int b = blockIdx.x, tid = threadIdx.x;
    if (tid == 0) {
        float a = b_log[0];
#pragma unroll
        for (int j = 0; j < N_RAIN; ++j) a += x[(size_t)b * N_FEAT + j] * w_log[j];
        lg_sh = sigf(a);
    }
    __syncthreads();
    const float lg = lg_sh;
    for (int i = tid; i < D_MODEL; i += 256)
        fused[i] = h[(size_t)b * D_MODEL + i] * (lg * W_gate[i] + b_gate[i]);
    __syncthreads();

    const int j = tid >> 3, r = tid & 7;           // 32 groups x 8 lanes
    float p = 0.0f;
    for (int d = r; d < D_MODEL; d += 8) p += fused[d] * W_h1[(size_t)j * D_MODEL + d];
    p += __shfl_xor(p, 4, 8);
    p += __shfl_xor(p, 2, 8);
    p += __shfl_xor(p, 1, 8);
    if (r == 0) z1[j] = fmaxf(p + b_h1[j], 0.0f);
    __syncthreads();
    if (tid == 0) {
        float logit = b_h2[0];
#pragma unroll
        for (int k = 0; k < 32; ++k) logit += z1[k] * W_h2[k];
        out[b] = sigf(logit);
    }
}

// ---------------------------------------------------------------------------
extern "C" void kernel_launch(void* const* d_in, const int* in_sizes, int n_in,
                              void* d_out, int out_size, void* d_ws, size_t ws_size,
                              hipStream_t stream)
{
    const float* x        = (const float*)d_in[0];
    const float* w_in     = (const float*)d_in[1];
    const float* b_in     = (const float*)d_in[2];
    const float* ln_g     = (const float*)d_in[3];
    const float* ln_b     = (const float*)d_in[4];
    const float* W_inproj = (const float*)d_in[5];
    const float* conv_w   = (const float*)d_in[6];
    const float* conv_b   = (const float*)d_in[7];
    const float* W_xproj  = (const float*)d_in[8];
    const float* W_dt     = (const float*)d_in[9];
    const float* b_dt     = (const float*)d_in[10];
    // d_in[11] = A_log : mathematically unused (h0 == 0, L == 1)
    const float* D_skip   = (const float*)d_in[12];
    const float* W_outproj= (const float*)d_in[13];
    const float* w_log    = (const float*)d_in[14];
    const float* b_log    = (const float*)d_in[15];
    const float* W_gate   = (const float*)d_in[16];
    const float* b_gate   = (const float*)d_in[17];
    const float* W_h1     = (const float*)d_in[18];
    const float* b_h1     = (const float*)d_in[19];
    const float* W_h2     = (const float*)d_in[20];
    const float* b_h2     = (const float*)d_in[21];
    (void)in_sizes; (void)n_in; (void)out_size; (void)ws_size;

    char* wsp = (char*)d_ws;
    size_t off = 0;
    auto alloc = [&](size_t bytes) -> void* {
        void* p = wsp + off;
        off += (bytes + 255) & ~(size_t)255;
        return p;
    };
    float*  h    = (float*)  alloc((size_t)BATCH * D_MODEL * 4);
    bf16_t* hn   = (bf16_t*) alloc((size_t)BATCH * D_MODEL * 2);
    float*  xz   = (float*)  alloc((size_t)BATCH * 2 * D_INNER * 4);
    float*  uf   = (float*)  alloc((size_t)BATCH * D_INNER * 4);
    bf16_t* ub   = (bf16_t*) alloc((size_t)BATCH * D_INNER * 2);
    float*  xdbl = (float*)  alloc((size_t)BATCH * XDBL * 4);
    bf16_t* yb   = (bf16_t*) alloc((size_t)BATCH * D_INNER * 2);
    bf16_t* wb   = (bf16_t*) alloc((size_t)2 * D_INNER * D_MODEL * 2); // largest weight

    head_in<<<BATCH, 256, 0, stream>>>(x, w_in, b_in, h);

    for (int l = 0; l < N_LAYERS; ++l) {
        layernorm_bf16<<<BATCH / 8, 256, 0, stream>>>(
            h, ln_g + (size_t)l * D_MODEL, ln_b + (size_t)l * D_MODEL, hn);

        { const int n = 2 * D_INNER * D_MODEL;   // 512 -> 3072 projection
          cvt_f32_bf16<<<n / 1024, 256, 0, stream>>>(W_inproj + (size_t)l * n, wb, n);
          gemm_bf16_wmma<4, false><<<dim3((2 * D_INNER) / 64, BATCH / 128), 256, 0, stream>>>(
              hn, wb, nullptr, xz, 2 * D_INNER, D_MODEL); }

        conv_silu<<<(BATCH * D_INNER) / 256, 256, 0, stream>>>(
            xz, conv_w + (size_t)l * D_INNER * 4, conv_b + (size_t)l * D_INNER, uf, ub);

        { const int n = XDBL * D_INNER;          // 1536 -> 96 projection (one 96-col block)
          cvt_f32_bf16<<<n / 1024, 256, 0, stream>>>(W_xproj + (size_t)l * n, wb, n);
          gemm_bf16_wmma<6, false><<<dim3(1, BATCH / 128), 256, 0, stream>>>(
              ub, wb, nullptr, xdbl, XDBL, D_INNER); }

        ssm_fused<<<dim3(D_INNER / 256, BATCH), 256, 0, stream>>>(
            xdbl, W_dt + (size_t)l * D_INNER * DT_RANK, b_dt + (size_t)l * D_INNER,
            D_skip + (size_t)l * D_INNER, uf, xz, yb);

        { const int n = D_MODEL * D_INNER;       // 1536 -> 512 projection (+residual)
          cvt_f32_bf16<<<n / 1024, 256, 0, stream>>>(W_outproj + (size_t)l * n, wb, n);
          gemm_bf16_wmma<4, true><<<dim3(D_MODEL / 64, BATCH / 128), 256, 0, stream>>>(
              yb, wb, h, h, D_MODEL, D_INNER); }
    }

    head_out<<<BATCH, 256, 0, stream>>>(h, x, w_log, b_log, W_gate, b_gate,
                                        W_h1, b_h1, W_h2, b_h2, (float*)d_out);
}
